// Mamba_82008105550545
// MI455X (gfx1250) — matrix operand order repaired and checked by reading, not compiled, not run on hardware
//
#include <hip/hip_runtime.h>
#include <hip/hip_bf16.h>

// ---------------------------------------------------------------------------
// Mamba block for MI455X (gfx1250, wave32, WMMA).
// Pipeline (all row-major (b*L, channels) -> zero transposes):
//   1. wmma_gemm      : xz = hidden @ W_in^T            (8192 x 4096 x 1024)
//   2. conv_silu_gate : xc = g0*silu(dwconv1)+g1*silu(dwconv2)
//   3. wmma_gemm      : x_dbl = xc @ W_x^T              (8192 x   96 x 2048)
//   4. wmma_gemm<SP>  : delta = softplus(x_dbl[:,:64] @ W_dt^T + b_dt)
//   5. scan_kernel    : selective scan, fused +D*x and *silu(z); y overwrites delta
//   6. wmma_gemm      : out = y @ W_out^T               (8192 x 1024 x 2048)
// GEMMs: bf16 WMMA (v_wmma_f32_16x16x32_bf16), f32 accumulate, double-buffered
// LDS staging with one barrier per K-tile so global_load_b128 latency overlaps
// the WMMA block.
// ---------------------------------------------------------------------------

typedef __attribute__((ext_vector_type(8)))  float   v8f;
typedef __attribute__((ext_vector_type(16))) __bf16  v16bf;

union BFragU { v16bf v; unsigned short u[16]; };

static __device__ __forceinline__ unsigned short f2bf(float f) {
    // native f32 -> bf16 convert (compiler picks the hardware cvt / packed cvt)
    __bf16 h = (__bf16)f;
    return __builtin_bit_cast(unsigned short, h);
}

// ---------------------------------------------------------------------------
// Generic  C[M,N] = A[M,K] (row-major, lda) @ B[N,K]^T (row-major, ldb)
// EPI==0: plain f32 store.  EPI==1: v = softplus(v + bias[col]).
// GUARD: bounds-check N (only needed for the N=96 GEMM; M,K always tile-exact).
// Tile 128x128x32, 8 waves, double-buffered LDS.
// ---------------------------------------------------------------------------
template <int EPI, bool GUARD>
__global__ __launch_bounds__(256) void wmma_gemm(
    const float* __restrict__ A, int lda,
    const float* __restrict__ B, int ldb,
    float* __restrict__ C, int ldc,
    const float* __restrict__ bias,
    int M, int N, int K)
{
    constexpr int BM = 128, BN = 128, BK = 32;
    constexpr int AST = 40;   // A LDS stride in halves (conflict-free frag reads)
    constexpr int BST = 136;  // B LDS stride in halves (K-major: [BK][BN+8])

    __shared__ unsigned short As[2][BM * AST];
    __shared__ unsigned short Bs[2][BK * BST];

    const int tid  = threadIdx.x;
    const int wave = tid >> 5;
    const int lane = tid & 31;
    const int lrow = lane & 15;
    const int khalf = (lane >> 4) * 8;      // A-frag K sub-offset
    const int wm = (wave & 3) * 32;         // 4 wave rows  -> 128
    const int wn = (wave >> 2) * 64;        // 2 wave cols  -> 128
    const int bm = blockIdx.y * BM;
    const int bn = blockIdx.x * BN;

    v8f acc[2][4];
#pragma unroll
    for (int mi = 0; mi < 2; ++mi)
#pragma unroll
        for (int ni = 0; ni < 4; ++ni)
            acc[mi][ni] = (v8f){0.f,0.f,0.f,0.f,0.f,0.f,0.f,0.f};

    float4 rA[4], rB[4];

    // issue global loads for one K-tile into registers
    auto gload = [&](int k0) {
#pragma unroll
        for (int i = 0; i < 4; ++i) {
            int slot = tid + i * 256;           // 1024 float4 slots per tile
            int r  = slot >> 3;
            int c4 = (slot & 7) * 4;
            if (GUARD) {
                rA[i] = make_float4(0.f, 0.f, 0.f, 0.f);
                rB[i] = make_float4(0.f, 0.f, 0.f, 0.f);
                if (bm + r < M)
                    rA[i] = *(const float4*)(A + (size_t)(bm + r) * lda + k0 + c4);
                if (bn + r < N)
                    rB[i] = *(const float4*)(B + (size_t)(bn + r) * ldb + k0 + c4);
            } else {
                rA[i] = *(const float4*)(A + (size_t)(bm + r) * lda + k0 + c4);
                rB[i] = *(const float4*)(B + (size_t)(bn + r) * ldb + k0 + c4);
            }
        }
    };

    // convert + store registers into LDS buffer p (A row-major, B K-major)
    auto stage = [&](int p) {
#pragma unroll
        for (int i = 0; i < 4; ++i) {
            int slot = tid + i * 256;
            int r  = slot >> 3;
            int c4 = (slot & 7) * 4;
            unsigned short* da = &As[p][r * AST + c4];
            da[0] = f2bf(rA[i].x); da[1] = f2bf(rA[i].y);
            da[2] = f2bf(rA[i].z); da[3] = f2bf(rA[i].w);
            unsigned short* db = &Bs[p][c4 * BST + r];
            db[0 * BST] = f2bf(rB[i].x); db[1 * BST] = f2bf(rB[i].y);
            db[2 * BST] = f2bf(rB[i].z); db[3 * BST] = f2bf(rB[i].w);
        }
    };

    gload(0);
    int p = 0;
    for (int k0 = 0; k0 < K; k0 += BK) {
        stage(p);
        __syncthreads();

        // prefetch next tile's globals while this tile's WMMAs run
        if (k0 + BK < K) gload(k0 + BK);

        // ---- load fragments from buffer p ----
        BFragU af[2], bfr[4];
#pragma unroll
        for (int mi = 0; mi < 2; ++mi) {
            const unsigned short* q = &As[p][(wm + mi * 16 + lrow) * AST];
#pragma unroll
            for (int j = 0; j < 8; ++j) af[mi].u[j]     = q[khalf + j];
#pragma unroll
            for (int j = 0; j < 8; ++j) af[mi].u[8 + j] = q[16 + khalf + j];
        }
#pragma unroll
        for (int ni = 0; ni < 4; ++ni) {
            // B frag: N = lane%16, K = table order (khalf + j, 16 + khalf + j)
            const unsigned short* q = &Bs[p][wn + ni * 16 + lrow];
#pragma unroll
            for (int j = 0; j < 8; ++j) bfr[ni].u[j] = q[(khalf + j) * BST];
#pragma unroll
            for (int j = 0; j < 8; ++j) bfr[ni].u[8 + j] = q[(16 + khalf + j) * BST];
        }

        // ---- 8 WMMAs ----
#pragma unroll
        for (int mi = 0; mi < 2; ++mi)
#pragma unroll
            for (int ni = 0; ni < 4; ++ni)
                acc[mi][ni] = __builtin_amdgcn_wmma_f32_16x16x32_bf16(
                    false, af[mi].v, false, bfr[ni].v,
                    (short)0, acc[mi][ni], false, false);

        p ^= 1;
    }

    // ---- epilogue / store ----
    const int half8 = (lane >> 4) * 8;
#pragma unroll
    for (int mi = 0; mi < 2; ++mi) {
#pragma unroll
        for (int ni = 0; ni < 4; ++ni) {
            int col = bn + wn + ni * 16 + lrow;
            if (GUARD && col >= N) continue;
            float bv = 0.f;
            if (EPI == 1) bv = bias[col];
#pragma unroll
            for (int r = 0; r < 8; ++r) {
                int row = bm + wm + mi * 16 + half8 + r;
                float v = acc[mi][ni][r];
                if (EPI == 1) {
                    v += bv;                                 // + b_dt
                    v = (v > 20.f) ? v : log1pf(__expf(v));  // softplus
                }
                C[(size_t)row * ldc + col] = v;
            }
        }
    }
}

// ---------------------------------------------------------------------------
// Dual causal depthwise conv (k=4, dil=1 and dil=2) + SiLU + softmax gate mix.
// xz layout: (b*L, 4096); x is cols [0,2048). Output xc: (b*L, 2048).
// ---------------------------------------------------------------------------
__global__ __launch_bounds__(256) void conv_silu_gate(
    const float* __restrict__ xz,
    const float* __restrict__ w1, const float* __restrict__ b1,
    const float* __restrict__ w2, const float* __restrict__ b2,
    const float* __restrict__ gates,
    float* __restrict__ xc, int L)
{
    constexpr int D = 2048;
    const int idx = blockIdx.x * blockDim.x + threadIdx.x; // over b*L*D
    const int d  = idx & (D - 1);
    const int bl = idx >> 11;          // b*L + l
    const int l  = bl & (L - 1);

    float g0 = gates[0], g1 = gates[1];
    float m  = fmaxf(g0, g1);
    float e0 = __expf(g0 - m), e1 = __expf(g1 - m);
    float inv = 1.f / (e0 + e1);
    g0 = e0 * inv; g1 = e1 * inv;

    float a1 = b1[d];
    float a2 = b2[d];
#pragma unroll
    for (int i = 0; i < 4; ++i) {
        int j1 = 3 - i;            // dil 1 tap offset
        int j2 = 2 * (3 - i);      // dil 2 tap offset
        if (l - j1 >= 0)
            a1 = fmaf(w1[d * 4 + i], xz[(size_t)(bl - j1) * 4096 + d], a1);
        if (l - j2 >= 0)
            a2 = fmaf(w2[d * 4 + i], xz[(size_t)(bl - j2) * 4096 + d], a2);
    }
    float s1 = a1 / (1.f + __expf(-a1));   // silu
    float s2 = a2 / (1.f + __expf(-a2));
    xc[(size_t)bl * D + d] = g0 * s1 + g1 * s2;
}

// ---------------------------------------------------------------------------
// Selective scan: one lane per (b, d) channel, h[16] in registers, serial in L.
// B_t/C_t (shared across d) staged through LDS in 64-step chunks.
// Fuses y += D*x and y *= silu(z). y overwrites the delta buffer in place.
// ---------------------------------------------------------------------------
__global__ __launch_bounds__(256) void scan_kernel(
    const float* __restrict__ xdbl,     // (b*L, 96): [64:80)=B, [80:96)=C
    const float* __restrict__ xc,       // (b*L, 2048)
    const float* __restrict__ xz,       // (b*L, 4096): z = cols [2048,4096)
    const float* __restrict__ A_log,    // (2048, 16)
    const float* __restrict__ D_param,  // (2048,)
    float* __restrict__ delta_y,        // in: delta, out: y   (b*L, 2048)
    int L)
{
    constexpr int D = 2048, DS = 16, TCH = 64;
    __shared__ float bc[TCH * 32];      // per-t: [0:16)=B, [16:32)=C

    const int tid = threadIdx.x;
    const int b   = blockIdx.x >> 3;                 // 8 blocks per batch
    const int d   = ((blockIdx.x & 7) << 8) + tid;

    float Av[DS];
#pragma unroll
    for (int n = 0; n < DS; ++n) Av[n] = -__expf(A_log[d * DS + n]);
    const float Dp = D_param[d];

    float h[DS];
#pragma unroll
    for (int n = 0; n < DS; ++n) h[n] = 0.f;

    for (int tc = 0; tc < L; tc += TCH) {
        // stage B/C chunk: 64 timesteps x 32 floats
#pragma unroll
        for (int i = 0; i < 8; ++i) {
            int s  = tid + i * 256;
            int tl = s >> 5, n = s & 31;
            bc[s] = xdbl[((size_t)(b * L + tc + tl)) * 96 + 64 + n];
        }
        __syncthreads();

        for (int tt = 0; tt < TCH; ++tt) {
            size_t bl  = (size_t)b * L + tc + tt;
            float dlt  = delta_y[bl * D + d];
            float xv   = xc[bl * D + d];
            float zv   = xz[bl * (size_t)(2 * D) + D + d];
            const float* Bv = &bc[tt * 32];
            const float* Cv = Bv + 16;
            float dx = dlt * xv;
            float y = 0.f;
#pragma unroll
            for (int n = 0; n < DS; ++n) {
                float hn = fmaf(h[n], __expf(dlt * Av[n]), dx * Bv[n]);
                h[n] = hn;
                y = fmaf(hn, Cv[n], y);
            }
            y = fmaf(Dp, xv, y);
            float sg = zv / (1.f + __expf(-zv));     // silu(z)
            delta_y[bl * D + d] = y * sg;
        }
        __syncthreads();
    }
}

// ---------------------------------------------------------------------------
extern "C" void kernel_launch(void* const* d_in, const int* in_sizes, int n_in,
                              void* d_out, int out_size, void* d_ws, size_t ws_size,
                              hipStream_t stream)
{
    (void)in_sizes; (void)n_in; (void)out_size; (void)ws_size;
    constexpr int Bsz = 4, L = 2048, DM = 1024, DI = 2048;
    constexpr int M = Bsz * L;                 // 8192

    const float* hidden  = (const float*)d_in[0];
    const float* W_in    = (const float*)d_in[1];   // (4096, 1024)
    const float* conv_w1 = (const float*)d_in[2];
    const float* conv_b1 = (const float*)d_in[3];
    const float* conv_w2 = (const float*)d_in[4];
    const float* conv_b2 = (const float*)d_in[5];
    const float* gates   = (const float*)d_in[6];
    const float* W_x     = (const float*)d_in[7];   // (96, 2048)
    const float* W_dt    = (const float*)d_in[8];   // (2048, 64)
    const float* b_dt    = (const float*)d_in[9];
    const float* A_log   = (const float*)d_in[10];
    const float* D_par   = (const float*)d_in[11];
    const float* W_out   = (const float*)d_in[12];  // (1024, 2048)
    float* out = (float*)d_out;

    // workspace carve-up (~259 MB)
    float* xz    = (float*)d_ws;                 // (8192, 4096)
    float* xc    = xz   + (size_t)M * (2 * DI);  // (8192, 2048)
    float* xdbl  = xc   + (size_t)M * DI;        // (8192, 96)
    float* delta = xdbl + (size_t)M * 96;        // (8192, 2048), becomes y

    dim3 blk(256);

    // 1) xz = hidden @ W_in^T   (tile-exact -> unguarded)
    wmma_gemm<0, false><<<dim3((2 * DI) / 128, M / 128), blk, 0, stream>>>(
        hidden, DM, W_in, DM, xz, 2 * DI, nullptr, M, 2 * DI, DM);

    // 2) conv + silu + gate mix
    conv_silu_gate<<<(M * DI) / 256, blk, 0, stream>>>(
        xz, conv_w1, conv_b1, conv_w2, conv_b2, gates, xc, L);

    // 3) x_dbl = xc @ W_x^T   (N = 96 -> guarded)
    wmma_gemm<0, true><<<dim3(1, M / 128), blk, 0, stream>>>(
        xc, DI, W_x, DI, xdbl, 96, nullptr, M, 96, DI);

    // 4) delta = softplus(x_dbl[:, :64] @ W_dt^T + b_dt)   (tile-exact)
    wmma_gemm<1, false><<<dim3(DI / 128, M / 128), blk, 0, stream>>>(
        xdbl, 96, W_dt, 64, delta, DI, b_dt, M, DI, 64);

    // 5) selective scan (y overwrites delta)
    scan_kernel<<<(Bsz * DI) / 256, blk, 0, stream>>>(
        xdbl, xc, xz, A_log, D_par, delta, L);

    // 6) out = y @ W_out^T   (tile-exact)
    wmma_gemm<0, false><<<dim3(DM / 128, M / 128), blk, 0, stream>>>(
        delta, DI, W_out, DI, out, DM, nullptr, M, DM, DI);
}
